// MambaModel_22076131901479
// MI455X (gfx1250) — compile-verified, matrix operand order checked by reading, not run on hardware
//
#include <hip/hip_runtime.h>
#include <math.h>

// ---------------- model constants ----------------
#define BATCH   4
#define SEQ     2048
#define DM      1024
#define DI      2048
#define NST     16
#define DTR     64
#define DCONV   4
#define XDBL_N  (DTR + 2 * NST)   // 96
#define MROWS   (BATCH * SEQ)     // 8192

typedef __attribute__((ext_vector_type(16))) __bf16 v16bf;
typedef __attribute__((ext_vector_type(8)))  float  v8f;

struct Frag { union { v16bf v; uint4 q[2]; }; };

// Per-lane WMMA A/B fragment load (16x32 / 32x16 bf16), row-major, ld in elems.
// lane l: row = l&15 ; K chunks at k0+8*(l>>4) and +16 (two 16B loads).
__device__ __forceinline__ void load_frag(Frag& f, const __bf16* __restrict__ base,
                                          long long row, long long ld, long long k) {
    const __bf16* p = base + row * ld + k;
    f.q[0] = *reinterpret_cast<const uint4*>(p);
    f.q[1] = *reinterpret_cast<const uint4*>(p + 16);
}

// ---------------- WMMA GEMM: C[M,N] = A[M,K] * W[N,K]^T (+bias, +softplus) ----
// A, W bf16 row-major; C f32 row-major.
// block = 256 thr = 8 waves (4 M x 2 N), block tile 128x128.
// wave tile 32x64 = 2x4 v_wmma_f32_16x16x32_bf16 (8 WMMA : 12 b128 loads / K-step).
// mode: 0 = none, 1 = +bias, 2 = softplus(x + bias)
__global__ __launch_bounds__(256) void k_wmma_gemm(
    const __bf16* __restrict__ A, const __bf16* __restrict__ W,
    float* __restrict__ C, int M, int N, int K,
    const float* __restrict__ bias, int mode)
{
    const int lane = threadIdx.x & 31;
    const int wave = threadIdx.x >> 5;
    const int wm = wave >> 1;          // 0..3
    const int wn = wave & 1;           // 0..1
    const long long M0 = (long long)blockIdx.y * 128 + wm * 32;
    const long long N0 = (long long)blockIdx.x * 128 + wn * 64;
    const int rA  = lane & 15;
    const int kof = (lane >> 4) << 3;  // 0 or 8

    // clamp B rows once (cols >= N are discarded at store)
    long long brows[4];
#pragma unroll
    for (int j = 0; j < 4; ++j) {
        long long col = N0 + 16 * j + rA;
        brows[j] = (col >= N) ? (N - 1) : col;
    }

    v8f acc[2][4] = {};
    for (int k0 = 0; k0 < K; k0 += 32) {
        // prefetch the streams a few K-steps ahead (global_prefetch_b8)
        if (lane == 0 && k0 + 128 < K) {
            __builtin_prefetch(A + (M0 + rA) * (long long)K + k0 + 128, 0, 1);
            __builtin_prefetch(W + brows[0] * (long long)K + k0 + 128, 0, 1);
        }
        Frag fa[2], fb[4];
#pragma unroll
        for (int i = 0; i < 2; ++i)
            load_frag(fa[i], A, M0 + 16 * i + rA, K, k0 + kof);
#pragma unroll
        for (int j = 0; j < 4; ++j)
            load_frag(fb[j], W, brows[j], K, k0 + kof);
#pragma unroll
        for (int i = 0; i < 2; ++i)
#pragma unroll
            for (int j = 0; j < 4; ++j)
                acc[i][j] = __builtin_amdgcn_wmma_f32_16x16x32_bf16(
                    false, fa[i].v, false, fb[j].v, (short)0, acc[i][j], false, false);
    }

    // D/C layout: lane l -> col l&15 ; VGPR v -> row v + 8*(l>>4)
    const int colL = lane & 15;
    const int rowH = (lane >> 4) * 8;
#pragma unroll
    for (int i = 0; i < 2; ++i)
#pragma unroll
        for (int j = 0; j < 4; ++j) {
            long long col = N0 + 16 * j + colL;
            if (col >= N) continue;
            float bv = (mode != 0 && bias) ? bias[col] : 0.f;
            long long rbase = M0 + 16 * i + rowH;
#pragma unroll
            for (int v = 0; v < 8; ++v) {
                float val = acc[i][j][v] + bv;
                if (mode == 2) val = (val > 20.f) ? val : log1pf(__expf(val));
                C[(rbase + v) * (long long)N + col] = val;
            }
        }
}

// ---------------- elementwise / small kernels --------------------------------
__global__ void k_f32_to_bf16(const float* __restrict__ s, __bf16* __restrict__ d, long long n) {
    long long i = (long long)blockIdx.x * blockDim.x + threadIdx.x;
    if (i < n) d[i] = (__bf16)s[i];
}

__global__ void k_f32_to_bf16_2d(const float* __restrict__ s, long long sld,
                                 __bf16* __restrict__ d, long long dld,
                                 long long rows, long long cols) {
    long long i = (long long)blockIdx.x * blockDim.x + threadIdx.x;
    if (i >= rows * cols) return;
    long long r = i / cols, c = i - r * cols;
    d[r * dld + c] = (__bf16)s[r * sld + c];
}

// c[b,m] = leakyrelu(cond[b,:]) @ cond_w[m,:] + cond_b[m]
__global__ void k_cond(const float* __restrict__ cond, const float* __restrict__ cw,
                       const float* __restrict__ cb, float* __restrict__ c,
                       int B, int Dmo, int Dc) {
    int i = blockIdx.x * blockDim.x + threadIdx.x;
    if (i >= B * Dmo) return;
    int m = i % Dmo, b = i / Dmo;
    const float* cr = cond + (long long)b * Dc;
    const float* wr = cw + (long long)m * Dc;
    float acc = cb[m];
    for (int k = 0; k < Dc; ++k) {
        float v = cr[k];
        v = v >= 0.f ? v : 0.01f * v;
        acc += v * wr[k];
    }
    c[i] = acc;
}

// x[b,l,m] = pe[l,m] + c[b,m]
__global__ void k_addpe(const float* __restrict__ pe, const float* __restrict__ c,
                        float* __restrict__ x, long long total) {
    long long idx = (long long)blockIdx.x * blockDim.x + threadIdx.x;
    if (idx >= total) return;
    int m = (int)(idx % DM);
    long long r = idx / DM;
    int l = (int)(r % SEQ);
    int b = (int)(r / SEQ);
    x[idx] = pe[(long long)l * DM + m] + c[(long long)b * DM + m];
}

// causal depthwise conv(width 4) + silu. xz row-major [MROWS, 2*DI], xc half = cols [0,DI)
__global__ void k_conv_silu(const float* __restrict__ xz, const float* __restrict__ w,
                            const float* __restrict__ bias, float* __restrict__ xc,
                            long long total) {
    long long idx = (long long)blockIdx.x * blockDim.x + threadIdx.x;
    if (idx >= total) return;
    int d = (int)(idx % DI);
    long long r = idx / DI;          // b*SEQ + l
    int l = (int)(r % SEQ);
    float acc = bias[d];
#pragma unroll
    for (int j = 0; j < DCONV; ++j) {
        int back = (DCONV - 1) - j;  // taps at l-3+j
        if (l - back >= 0)
            acc += w[d * DCONV + j] * xz[(r - back) * (2LL * DI) + d];
    }
    xc[idx] = acc / (1.f + __expf(-acc));   // silu
}

// y *= silu(z), z = xz[:, DI + d]
__global__ void k_gate(float* __restrict__ y, const float* __restrict__ xz, long long total) {
    long long idx = (long long)blockIdx.x * blockDim.x + threadIdx.x;
    if (idx >= total) return;
    int d = (int)(idx % DI);
    long long r = idx / DI;
    float z = xz[r * (2LL * DI) + DI + d];
    y[idx] *= z / (1.f + __expf(-z));
}

// ---------------- selective scan ---------------------------------------------
// one lane per (b,d) channel, 16 states in registers; B/C rows staged in LDS.
#define SCAN_T 64
__global__ __launch_bounds__(256) void k_scan(
    const float* __restrict__ delta, const float* __restrict__ u,
    const float* __restrict__ xdbl, const float* __restrict__ A_log,
    const float* __restrict__ Dp, float* __restrict__ y)
{
    __shared__ float sB[SCAN_T * NST];
    __shared__ float sC[SCAN_T * NST];
    const int tid = threadIdx.x;
    const int blocksPerB = DI / 256;
    const int b    = blockIdx.x / blocksPerB;
    const int d    = (blockIdx.x % blocksPerB) * 256 + tid;

    float Av[NST], h[NST];
#pragma unroll
    for (int n = 0; n < NST; ++n) {
        Av[n] = -__expf(A_log[(long long)d * NST + n]);
        h[n]  = 0.f;
    }
    const float Dd = Dp[d];

    for (int l0 = 0; l0 < SEQ; l0 += SCAN_T) {
        __syncthreads();
        for (int e = tid; e < SCAN_T * NST; e += 256) {
            int t = e >> 4, n = e & 15;
            long long base = ((long long)b * SEQ + l0 + t) * XDBL_N;
            sB[e] = xdbl[base + DTR + n];
            sC[e] = xdbl[base + DTR + NST + n];
        }
        __syncthreads();
        for (int t = 0; t < SCAN_T; ++t) {
            long long idx = ((long long)b * SEQ + l0 + t) * DI + d;  // lanes: consecutive d -> coalesced
            float dl = delta[idx];
            float uu = u[idx];
            float du = dl * uu;
            float acc = 0.f;
#pragma unroll
            for (int n = 0; n < NST; ++n) {
                float dA = __expf(dl * Av[n]);
                h[n] = h[n] * dA + du * sB[(t << 4) + n];
                acc += h[n] * sC[(t << 4) + n];
            }
            y[idx] = acc + uu * Dd;
        }
    }
}

// ---------------- host orchestration ----------------------------------------
static inline int cdiv(long long a, long long b) { return (int)((a + b - 1) / b); }

extern "C" void kernel_launch(void* const* d_in, const int* in_sizes, int n_in,
                              void* d_out, int out_size, void* d_ws, size_t ws_size,
                              hipStream_t stream) {
    const float* pe        = (const float*)d_in[0];
    const float* condition = (const float*)d_in[1];
    const float* cond_w    = (const float*)d_in[2];
    const float* cond_b    = (const float*)d_in[3];
    // d_in[4] = output_shape0 (unused: B==4 path, no tiling branch)

    // workspace layout
    char* ws = (char*)d_ws;
    size_t off = 0;
    auto take = [&](size_t bytes) -> void* {
        void* p = ws + off;
        off += (bytes + 255) & ~(size_t)255;
        return p;
    };
    float*  cbuf   = (float*) take((size_t)BATCH * DM * 4);
    float*  xbuf   = (float*) take((size_t)MROWS * DM * 4);
    float*  xz     = (float*) take((size_t)MROWS * 2 * DI * 4);
    float*  xc     = (float*) take((size_t)MROWS * DI * 4);
    float*  xdbl   = (float*) take((size_t)MROWS * XDBL_N * 4);
    float*  deltab = (float*) take((size_t)MROWS * DI * 4);
    float*  ybuf   = (float*) take((size_t)MROWS * DI * 4);
    __bf16* abf    = (__bf16*)take((size_t)MROWS * DI * 2);      // activation bf16 staging
    __bf16* dtbf   = (__bf16*)take((size_t)MROWS * DTR * 2);
    __bf16* inw_bf = (__bf16*)take((size_t)2 * DI * DM * 2);
    __bf16* xw_bf  = (__bf16*)take((size_t)XDBL_N * DI * 2);
    __bf16* dtw_bf = (__bf16*)take((size_t)DI * DTR * 2);
    __bf16* outw_bf= (__bf16*)take((size_t)DM * DI * 2);
    (void)ws_size; (void)in_sizes; (void)n_in; (void)out_size;

    const long long nXM  = (long long)MROWS * DM;
    const long long nXC  = (long long)MROWS * DI;

    auto cvt = [&](const float* s, __bf16* d, long long n) {
        k_f32_to_bf16<<<cdiv(n, 256), 256, 0, stream>>>(s, d, n);
    };
    auto gemm = [&](const __bf16* A, const __bf16* W, float* C, int N, int K,
                    const float* bias, int mode) {
        dim3 g(cdiv(N, 128), MROWS / 128);
        k_wmma_gemm<<<g, 256, 0, stream>>>(A, W, C, MROWS, N, K, bias, mode);
    };

    // x = pe + Linear(LeakyReLU(condition))
    k_cond<<<cdiv(BATCH * DM, 256), 256, 0, stream>>>(condition, cond_w, cond_b, cbuf,
                                                      BATCH, DM, DM);
    k_addpe<<<cdiv(nXM, 256), 256, 0, stream>>>(pe, cbuf, xbuf, nXM);

    auto run_block = [&](const float* xin, float* xout, int pbase) {
        const float* in_w   = (const float*)d_in[pbase + 0];
        const float* conv_w = (const float*)d_in[pbase + 1];
        const float* conv_b = (const float*)d_in[pbase + 2];
        const float* x_w    = (const float*)d_in[pbase + 3];
        const float* dt_w   = (const float*)d_in[pbase + 4];
        const float* dt_b   = (const float*)d_in[pbase + 5];
        const float* A_log  = (const float*)d_in[pbase + 6];
        const float* Dp     = (const float*)d_in[pbase + 7];
        const float* out_w  = (const float*)d_in[pbase + 8];

        // xz = x @ in_w^T         [8192 x 4096], K=1024
        cvt(xin, abf, nXM);
        cvt(in_w, inw_bf, (long long)2 * DI * DM);
        gemm(abf, inw_bf, xz, 2 * DI, DM, nullptr, 0);

        // xc = silu(causal_dwconv(xz[:, :DI]))
        k_conv_silu<<<cdiv(nXC, 256), 256, 0, stream>>>(xz, conv_w, conv_b, xc, nXC);

        // x_dbl = xc @ x_w^T      [8192 x 96], K=2048
        cvt(xc, abf, nXC);
        cvt(x_w, xw_bf, (long long)XDBL_N * DI);
        gemm(abf, xw_bf, xdbl, XDBL_N, DI, nullptr, 0);

        // delta = softplus(dt @ dt_w^T + dt_b)   [8192 x 2048], K=64
        k_f32_to_bf16_2d<<<cdiv((long long)MROWS * DTR, 256), 256, 0, stream>>>(
            xdbl, XDBL_N, dtbf, DTR, MROWS, DTR);
        cvt(dt_w, dtw_bf, (long long)DI * DTR);
        gemm(dtbf, dtw_bf, deltab, DI, DTR, dt_b, 2);

        // selective scan -> y (includes + xc*D skip)
        k_scan<<<BATCH * (DI / 256), 256, 0, stream>>>(deltab, xc, xdbl, A_log, Dp, ybuf);

        // y *= silu(z)
        k_gate<<<cdiv(nXC, 256), 256, 0, stream>>>(ybuf, xz, nXC);

        // out = y @ out_w^T       [8192 x 1024], K=2048
        cvt(ybuf, abf, nXC);
        cvt(out_w, outw_bf, (long long)DM * DI);
        gemm(abf, outw_bf, xout, DM, DI, nullptr, 0);
    };

    run_block(xbuf, xbuf, 5);                 // block 1 (in-place through xbuf)
    run_block(xbuf, (float*)d_out, 14);       // block 2 -> final output [4,2048,1024]
}
